// GraphConv_26603027431482
// MI455X (gfx1250) — matrix operand order repaired and checked by reading, not compiled, not run on hardware
//
#include <hip/hip_runtime.h>
#include <hip/hip_bf16.h>

typedef __attribute__((ext_vector_type(16))) _Float16 v16h;
typedef __attribute__((ext_vector_type(8)))  float    v8f;
typedef __attribute__((ext_vector_type(4)))  float    v4f;

#define NNODES 100000
#define NEDGES 625000
#define D      128
#define TILE_EDGES 128   // 8 waves x 16 edges per block

__global__ __launch_bounds__(256)
void graphconv_wmma_kernel(const float* __restrict__ x,
                           const int*   __restrict__ eidx,   // [2, NEDGES]
                           const float* __restrict__ W,      // [D, D] row-major (k, n)
                           const float* __restrict__ b,      // [D]
                           float*       __restrict__ out,    // [NEDGES, D]
                           int nTiles)
{
    // W staged as f16 in WMMA B-fragment layout: [kChunk][nTile][lane] -> 16 halves (32B)
    __shared__ v16h ldsW[4][8][32];   // 32 KB

    const int tid  = threadIdx.x;
    const int lane = tid & 31;
    const int wave = tid >> 5;

    // ---- one-time: pack W (f32 -> f16) into B-fragment layout ----
    // B 32x16 f16 fragment: lane -> n = nTile*16 + (lane&15);
    // halves h=0..7 -> K = kChunk*32 + ((lane>=16)?8:0) + h ; h=8..15 -> K = kChunk*32 + 16 + ((lane>=16)?8:0) + (h-8)
    for (int i = tid; i < 4 * 8 * 32; i += 256) {
        const int kk = i >> 8;
        const int nt = (i >> 5) & 7;
        const int l2 = i & 31;
        const int n  = nt * 16 + (l2 & 15);
        const int kb = (l2 >= 16) ? 8 : 0;
        v16h tmp;
        #pragma unroll
        for (int h = 0; h < 16; ++h) {
            const int k = kk * 32 + kb + (h < 8 ? h : 8 + h);
            tmp[h] = (_Float16)W[k * D + n];
        }
        ldsW[kk][nt][l2] = tmp;
    }
    __syncthreads();

    // Bias in C/D layout depends only on N = lane&15 (same for all 8 row-VGPRs)
    float bias[8];
    #pragma unroll
    for (int nt = 0; nt < 8; ++nt) bias[nt] = b[nt * 16 + (lane & 15)];

    // A 16x32 f16 fragment mapping: m = lane&15; kbase = (lane>=16)?8:0;
    // h=0..7 -> K = kk*32 + kbase + h ; h=8..15 -> K = kk*32 + 16 + kbase + (h-8)
    const int m  = lane & 15;
    const int kb = (lane >= 16) ? 8 : 0;

    for (int tile = blockIdx.x; tile < nTiles; tile += gridDim.x) {
        const int e0 = tile * TILE_EDGES + wave * 16;
        if (e0 >= NEDGES) continue;               // uniform per wave

        int e = e0 + m;
        if (e > NEDGES - 1) e = NEDGES - 1;       // clamp for tail (stores predicated)
        int r = eidx[e];
        int c = eidx[NEDGES + e];
        r = (r < 0) ? 0 : (r > NNODES - 1 ? NNODES - 1 : r);   // jnp.clip
        c = (c < 0) ? 0 : (c > NNODES - 1 ? NNODES - 1 : c);

        const v4f* xr = (const v4f*)(x + (size_t)r * D);
        const v4f* xc = (const v4f*)(x + (size_t)c * D);

        v8f acc[8];
        #pragma unroll
        for (int nt = 0; nt < 8; ++nt) {
            #pragma unroll
            for (int j = 0; j < 8; ++j) acc[nt][j] = bias[nt];
        }

        #pragma unroll
        for (int kk = 0; kk < 4; ++kk) {
            // run0: K = kk*32 + kb + 0..7 ; run1: K = kk*32 + 16 + kb + 0..7
            const int f0 = kk * 8 + (kb >> 2);    // float4 index of run0
            const v4f r0 = xr[f0],     r1 = xr[f0 + 1];
            const v4f r2 = xr[f0 + 4], r3 = xr[f0 + 5];
            const v4f c0 = xc[f0],     c1 = xc[f0 + 1];
            const v4f c2 = xc[f0 + 4], c3 = xc[f0 + 5];

            v16h a;
            #pragma unroll
            for (int j = 0; j < 4; ++j) {
                a[j]      = (_Float16)(r0[j] + c0[j]);
                a[4 + j]  = (_Float16)(r1[j] + c1[j]);
                a[8 + j]  = (_Float16)(r2[j] + c2[j]);
                a[12 + j] = (_Float16)(r3[j] + c3[j]);
            }

            #pragma unroll
            for (int nt = 0; nt < 8; ++nt) {
                const v16h bf = ldsW[kk][nt][lane];
                acc[nt] = __builtin_amdgcn_wmma_f32_16x16x32_f16(
                    false, a, false, bf, (short)0, acc[nt], false, false);
            }
        }

        // ---- fused ReLU + non-temporal store (output is write-once streaming
        // traffic; NT keeps the 51MB node table resident in the 192MB L2).
        // C/D layout: VGPR rr -> M = rr + (lane>=16 ? 8:0), N = lane&15
        const int n0  = lane & 15;
        const int mhi = (lane >= 16) ? 8 : 0;
        #pragma unroll
        for (int rr = 0; rr < 8; ++rr) {
            const int erow = e0 + mhi + rr;
            if (erow < NEDGES) {
                float* op = out + (size_t)erow * D + n0;
                #pragma unroll
                for (int nt = 0; nt < 8; ++nt) {
                    const float v = acc[nt][rr];
                    __builtin_nontemporal_store(v > 0.0f ? v : 0.0f, op + nt * 16);
                }
            }
        }
    }
}

extern "C" void kernel_launch(void* const* d_in, const int* in_sizes, int n_in,
                              void* d_out, int out_size, void* d_ws, size_t ws_size,
                              hipStream_t stream) {
    const float* x    = (const float*)d_in[0];
    const int*   eidx = (const int*)  d_in[1];
    const float* W    = (const float*)d_in[2];
    const float* b    = (const float*)d_in[3];
    float*       out  = (float*)d_out;

    const int nTiles = (NEDGES + TILE_EDGES - 1) / TILE_EDGES;   // 4883
    const int blocks = nTiles < 2048 ? nTiles : 2048;            // persistent, grid-stride

    graphconv_wmma_kernel<<<blocks, 256, 0, stream>>>(x, eidx, W, b, out, nTiles);
}
